// ScaledDotProductAttention_33569464386362
// MI455X (gfx1250) — matrix operand order repaired and checked by reading, not compiled, not run on hardware
//
#include <hip/hip_runtime.h>

typedef __attribute__((ext_vector_type(16))) _Float16 v16h;
typedef __attribute__((ext_vector_type(8)))  _Float16 h8;
typedef __attribute__((ext_vector_type(8)))  float    v8f;
typedef __attribute__((ext_vector_type(4)))  float    f4;

namespace {
constexpr int BATCH = 2;
constexpr int HEADS = 16;
constexpr int SEQ   = 2048;
constexpr int DKDIM = 128;
constexpr int BH    = BATCH * HEADS;          // 32
constexpr int QTILES = SEQ / 16;              // 128
constexpr float GAMMA_C = 0.7f;
constexpr float SCALE_C = 0.08838834764831845f; // 1/sqrt(128)

// dynamic LDS layout (bytes)
constexpr int OFF_Q   = 0;                        // 16*128 f16  = 4 KB
constexpr int OFF_SC  = OFF_Q  + 16 * DKDIM * 2;  // 16*2048 f32 = 128 KB
constexpr int OFF_BL  = OFF_SC + 16 * SEQ * 4;    // 16*2048 f16 = 64 KB
constexpr int OFF_RED = OFF_BL + 16 * SEQ * 2;    // 16*16 f32   = 1 KB
constexpr int LDS_BYTES = OFF_RED + 16 * 16 * 4;  // 201728 B (< 320 KB/WGP)
}

__global__ __launch_bounds__(256)
void attn_fused_kernel(const float* __restrict__ Qg,
                       const float* __restrict__ Kg,
                       const float* __restrict__ Vg,
                       const float* __restrict__ visMg,
                       float* __restrict__ ctxg,
                       float* __restrict__ attng)
{
    extern __shared__ char smem[];
    _Float16* Qs  = (_Float16*)(smem + OFF_Q);    // [16][128] f16
    float*    Sc  = (float*)   (smem + OFF_SC);   // [16][2048] f32 scores
    _Float16* Bl  = (_Float16*)(smem + OFF_BL);   // [16][2048] f16 blended
    float*    Red = (float*)   (smem + OFF_RED);  // [16][16] reduce scratch

    const int bh    = blockIdx.x / QTILES;
    const int qt    = blockIdx.x % QTILES;
    const int qbase = qt * 16;

    const int tid  = threadIdx.x;
    const int lane = tid & 31;        // wave32
    const int wave = tid >> 5;        // 8 waves
    const int nl   = lane & 15;       // row (A) / col (B,C) within 16x16
    const int hs   = lane >> 4;       // half-select per CDNA5 WMMA layout

    const size_t qoff = ((size_t)bh * SEQ + qbase) * DKDIM;

    // ---- Phase 0: stage Q tile (16x128) in LDS as f16 ----
    for (int i = tid; i < 16 * DKDIM; i += 256)
        Qs[i] = (_Float16)Qg[qoff + i];
    __syncthreads();

    // ---- Phase 1: scores = (Q K^T) * scale, 16x2048 into LDS ----
    // wave w handles k-tiles w, w+8, ... (16 tiles each); 4 wmma per tile (K=128)
    for (int kt = wave; kt < QTILES; kt += 8) {
        v8f c = {};
        const float* krow = Kg + ((size_t)bh * SEQ + kt * 16 + nl) * DKDIM;
        #pragma unroll
        for (int s = 0; s < 4; ++s) {
            // A fragment: contiguous k = 16*hs + e (matches B)
            v16h a;
            const h8 alo = *(const h8*)(Qs + nl * DKDIM + s * 32 + hs * 16);
            const h8 ahi = *(const h8*)(Qs + nl * DKDIM + s * 32 + hs * 16 + 8);
            #pragma unroll
            for (int i = 0; i < 8; ++i) { a[i] = alo[i]; a[i + 8] = ahi[i]; }
            // B fragment: B[d][n] = K[kcol=n][d]; lane reads 16 consecutive f32
            v16h b;
            const float* kp = krow + s * 32 + hs * 16;
            #pragma unroll
            for (int i = 0; i < 4; ++i) {
                f4 t = *(const f4*)(kp + i * 4);
                b[4 * i + 0] = (_Float16)t[0];
                b[4 * i + 1] = (_Float16)t[1];
                b[4 * i + 2] = (_Float16)t[2];
                b[4 * i + 3] = (_Float16)t[3];
            }
            c = __builtin_amdgcn_wmma_f32_16x16x32_f16(false, a, false, b,
                                                       (short)0, c, false, false);
        }
        #pragma unroll
        for (int r = 0; r < 8; ++r) {
            const int m = r + 8 * hs;                // C layout: M = vgpr + 8*hs
            Sc[m * SEQ + kt * 16 + nl] = c[r] * SCALE_C;
        }
    }
    __syncthreads();

    // ---- Phase 2: row softmax in f32; fused attn write + blended build ----
    const int row = tid >> 4;     // 0..15 query row
    const int ct  = tid & 15;     // 16 threads cooperate per row
    float* srow = Sc + row * SEQ;

    float mx = -3.4e38f;
    for (int j = ct; j < SEQ; j += 16) mx = fmaxf(mx, srow[j]);
    Red[row * 16 + ct] = mx;
    __syncthreads();
    float rowmax = Red[row * 16];
    #pragma unroll
    for (int i = 1; i < 16; ++i) rowmax = fmaxf(rowmax, Red[row * 16 + i]);
    __syncthreads();

    float sm = 0.f;
    for (int j = ct; j < SEQ; j += 16) {
        float e = __expf(srow[j] - rowmax);
        srow[j] = e;
        sm += e;
    }
    Red[row * 16 + ct] = sm;
    __syncthreads();
    float rowsum = 0.f;
    #pragma unroll
    for (int i = 0; i < 16; ++i) rowsum += Red[row * 16 + i];
    const float inv = 1.0f / rowsum;

    // single pass over the S-sized row: read visM once, write attn once
    const size_t arow = ((size_t)bh * SEQ + qbase + row) * SEQ;
    for (int j = ct; j < SEQ; j += 16) {
        const float a  = srow[j] * inv;
        attng[arow + j] = a;
        const float bld = a * GAMMA_C + visMg[arow + j] * (1.0f - GAMMA_C);
        Bl[row * SEQ + j] = (_Float16)bld;
    }
    __syncthreads();

    // ---- Phase 3: context = blended @ V ; wave owns 16 columns of DK ----
    const int col = wave * 16 + nl;
    v8f c2 = {};
    for (int s = 0; s < SEQ / 32; ++s) {
        v16h a;
        const h8 alo = *(const h8*)(Bl + nl * SEQ + s * 32 + hs * 16);
        const h8 ahi = *(const h8*)(Bl + nl * SEQ + s * 32 + hs * 16 + 8);
        #pragma unroll
        for (int i = 0; i < 8; ++i) { a[i] = alo[i]; a[i + 8] = ahi[i]; }
        // B fragment: B[k][n] = V[k][col]; per e, lanes 0-15 read 64B contiguous
        v16h b;
        const float* vp = Vg + ((size_t)bh * SEQ + s * 32 + hs * 16) * DKDIM + col;
        #pragma unroll
        for (int e = 0; e < 16; ++e) b[e] = (_Float16)vp[e * DKDIM];
        c2 = __builtin_amdgcn_wmma_f32_16x16x32_f16(false, a, false, b,
                                                    (short)0, c2, false, false);
    }
    #pragma unroll
    for (int r = 0; r < 8; ++r) {
        const int m = r + 8 * hs;
        ctxg[qoff + m * DKDIM + col] = c2[r];
    }
}

extern "C" void kernel_launch(void* const* d_in, const int* in_sizes, int n_in,
                              void* d_out, int out_size, void* d_ws, size_t ws_size,
                              hipStream_t stream) {
    (void)in_sizes; (void)n_in; (void)out_size; (void)d_ws; (void)ws_size;
    const float* Q    = (const float*)d_in[0];
    const float* K    = (const float*)d_in[1];
    const float* V    = (const float*)d_in[2];
    // d_in[3] = attn_mask (unused by reference forward)
    const float* visM = (const float*)d_in[4];

    float* ctx  = (float*)d_out;                                   // [B,H,S,DK]
    float* attn = ctx + (size_t)BATCH * HEADS * SEQ * DKDIM;       // [B,H,S,S]

    dim3 grid(BH * QTILES);   // 4096 workgroups, one per 16-row query tile
    dim3 block(256);          // 8 wave32 waves
    attn_fused_kernel<<<grid, block, LDS_BYTES, stream>>>(Q, K, V, visM, ctx, attn);
}
